// MambaLayer_26362509263195
// MI455X (gfx1250) — matrix operand order repaired and checked by reading, not compiled, not run on hardware
//
#include <hip/hip_runtime.h>
#include <hip/hip_bf16.h>
#include <math.h>

typedef __bf16 bf16_t;
typedef __attribute__((ext_vector_type(16))) __bf16 v16bf;
typedef __attribute__((ext_vector_type(8)))  __bf16 v8bf;
typedef __attribute__((ext_vector_type(8)))  float  v8f;

#define BATCH 4
#define LSEQ  2048
#define DMOD  1024
#define DSTATE 16
#define KCONV 4
#define DI    2048      // EXPAND * DM
#define NPROJ 4096      // 2*DI
#define NBC   32        // 2*DS

// ---------------------------------------------------------------- helpers
__device__ __forceinline__ unsigned short f2bf(float f) {
    union { float f; unsigned u; } v; v.f = f;
    unsigned r = v.u + 0x7FFFu + ((v.u >> 16) & 1u);   // round-to-nearest-even
    return (unsigned short)(r >> 16);
}

__global__ void cvt_f32_to_bf16(const float* __restrict__ in,
                                unsigned short* __restrict__ out, int n) {
    int i = blockIdx.x * blockDim.x + threadIdx.x;
    if (i < n) out[i] = f2bf(in[i]);
}

// ---------------------------------------------------------------- WMMA GEMM
// C(M,N) = A(M,K) @ B(N,K)^T + bias(N).  A,B bf16 row-major; out f32.
// K and N are compile-time: all tile strides fold into 24-bit instruction
// offsets, only two per-lane 64-bit base pointers live in the loop.
// One wave computes a 32(M) x (NT*16)(N) macro-tile -> 2*NT v_wmma / k-step.
// Copy-free ping-pong double buffering; "#pragma unroll 1" pins the pipeline
// depth at 2 so the compiler cannot re-unroll and blow the register file
// (the live set is ~64 acc + ~96 frag VGPRs); launch_bounds(256,1) gives the
// allocator the full per-wave VGPR budget -> no scratch spills.
// K/32 even, N % (NT*16) == 0, M % 32 == 0 for all three GEMMs: no guards ->
// EXEC is all-ones around every WMMA (ISA 7.12 requirement).
// SPLIT==1: columns [0,splitN) -> out0, [splitN,N) -> out1 (both M x splitN).
template<int NT, int SPLIT, int KK, int NN>
__global__ __launch_bounds__(256, 1)
void gemm_wmma_bf16(const bf16_t* __restrict__ A,
                    const bf16_t* __restrict__ B,
                    const float* __restrict__ bias,
                    float* __restrict__ out0,
                    float* __restrict__ out1,
                    int M, int splitN)
{
    const int lane = threadIdx.x & 31;
    const int wid  = threadIdx.x >> 5;
    constexpr int tilesN = NN / (NT * 16);
    const int wg = blockIdx.x * 8 + wid;
    const int tm = wg / tilesN;
    const int tn = wg % tilesN;
    if (tm * 32 >= M) return;                 // wave-uniform exit

    const int li = lane & 15;
    const int kA = (lane < 16) ? 0 : 8;       // A frag K-base per lane half
    const int kB = (lane < 16) ? 0 : 8;       // B frag K-base per lane half
    const int m0 = tm * 32;
    const int n0 = tn * (NT * 16);

    const bf16_t* aPtr = A + (size_t)(m0 + li) * KK + kA;  // A rows m0+li, m0+16+li
    const bf16_t* bPtr = B + (size_t)(n0 + li) * KK + kB;  // B rows n0+li (+t*16)

    // A 16x32 fragment: halves 0..7 = K kA..kA+7, halves 8..15 = K kA+16..23
    auto ldA = [&](int half, int k) -> v16bf {
        const bf16_t* p = aPtr + half * (16 * KK) + k;     // constant stride
        v8bf lo = *(const v8bf*)(p);
        v8bf hi = *(const v8bf*)(p + 16);
        return __builtin_shufflevector(lo, hi,
                   0,1,2,3,4,5,6,7,8,9,10,11,12,13,14,15);
    };
    // B 32x16 fragment: halves 0..7 = K kB..kB+7, halves 8..15 = K kB+16..23
    auto ldB = [&](int t, int k) -> v16bf {
        const bf16_t* p = bPtr + t * (16 * KK) + k;        // constant stride
        v8bf lo = *(const v8bf*)(p);
        v8bf hi = *(const v8bf*)(p + 16);
        return __builtin_shufflevector(lo, hi,
                   0,1,2,3,4,5,6,7,8,9,10,11,12,13,14,15);
    };

    v8f acc0[NT];
    v8f acc1[NT];
    #pragma unroll
    for (int t = 0; t < NT; ++t) { v8f z = {}; acc0[t] = z; acc1[t] = z; }

    v16bf a0_0, a1_0, b_0[NT];                 // ping buffer
    v16bf a0_1, a1_1, b_1[NT];                 // pong buffer

    auto mma = [&](v16bf a0, v16bf a1, v16bf (&bf)[NT]) {
        #pragma unroll
        for (int t = 0; t < NT; ++t) {
            acc0[t] = __builtin_amdgcn_wmma_f32_16x16x32_bf16(
                          false, a0, false, bf[t], (short)0, acc0[t], false, false);
            acc1[t] = __builtin_amdgcn_wmma_f32_16x16x32_bf16(
                          false, a1, false, bf[t], (short)0, acc1[t], false, false);
        }
    };

    // ---- prologue: fill both buffers
    a0_0 = ldA(0, 0);  a1_0 = ldA(1, 0);
    #pragma unroll
    for (int t = 0; t < NT; ++t) b_0[t] = ldB(t, 0);
    a0_1 = ldA(0, 32); a1_1 = ldA(1, 32);
    #pragma unroll
    for (int t = 0; t < NT; ++t) b_1[t] = ldB(t, 32);

    // ---- steady state: consume a buffer, immediately refill it (copy-free)
    #pragma unroll 1
    for (int k = 64; k < KK; k += 64) {
        mma(a0_0, a1_0, b_0);
        a0_0 = ldA(0, k); a1_0 = ldA(1, k);
        #pragma unroll
        for (int t = 0; t < NT; ++t) b_0[t] = ldB(t, k);
        __builtin_prefetch(aPtr + k + 512, 0, 3);              // stream A ahead
        __builtin_prefetch(aPtr + 16 * KK + k + 512, 0, 3);

        mma(a0_1, a1_1, b_1);
        a0_1 = ldA(0, k + 32); a1_1 = ldA(1, k + 32);
        #pragma unroll
        for (int t = 0; t < NT; ++t) b_1[t] = ldB(t, k + 32);
    }

    // ---- epilogue: drain both buffers
    mma(a0_0, a1_0, b_0);
    mma(a0_1, a1_1, b_1);

    // C/D layout: VGPR r holds M = r (lanes 0-15) or r+8 (lanes 16-31); N = lane&15
    #pragma unroll
    for (int mt = 0; mt < 2; ++mt) {
        const int rbase = m0 + mt * 16 + ((lane < 16) ? 0 : 8);
        #pragma unroll
        for (int t = 0; t < NT; ++t) {
            const int col = n0 + t * 16 + li;
            const float bv = bias[col];
            const v8f acc = mt ? acc1[t] : acc0[t];
            #pragma unroll
            for (int r = 0; r < 8; ++r) {
                const int row = rbase + r;
                const float v = acc[r] + bv;
                if (SPLIT) {
                    if (col < splitN) out0[(size_t)row * splitN + col] = v;
                    else              out1[(size_t)row * splitN + (col - splitN)] = v;
                } else {
                    out0[(size_t)row * NN + col] = v;
                }
            }
        }
    }
}

// ---------------------------------------------------------------- conv + silu
__global__ void conv_silu(const float* __restrict__ xi,
                          const float* __restrict__ conv_w,
                          const float* __restrict__ conv_b,
                          unsigned short* __restrict__ xsb)
{
    int idx = blockIdx.x * blockDim.x + threadIdx.x;
    if (idx >= BATCH * LSEQ * DI) return;
    int c = idx % DI;
    int l = (idx / DI) % LSEQ;
    int b = idx / (DI * LSEQ);
    float acc = conv_b[c];
    const float* w    = conv_w + c * KCONV;            // (DI,1,K)
    const float* xcol = xi + (size_t)b * LSEQ * DI + c;
    #pragma unroll
    for (int t = 0; t < KCONV; ++t) {
        int ll = l - (KCONV - 1) + t;                  // causal left pad
        if (ll >= 0) acc += w[t] * xcol[(size_t)ll * DI];
    }
    float s = acc / (1.f + __expf(-acc));              // silu
    xsb[idx] = f2bf(s);
}

// ---------------------------------------------------------------- scan
// h_new = tanh(exp(exp(dt_{l%16}) * A) @ h + B_t*h + C_t); one wave per batch.
// dt index cycles mod 16 -> only 16 distinct exp(exp(dt_k)*A) matrices: LDS.
__global__ void scan_kernel(const float* __restrict__ bc,   // (B*L, 32)
                            const float* __restrict__ dt,   // (DI,)
                            const float* __restrict__ A,    // (16,16)
                            float* __restrict__ hs)         // (B*L, 16)
{
    __shared__ float Ahat[DSTATE * DSTATE * DSTATE];   // 16 matrices of 16x16
    for (int idx = threadIdx.x; idx < DSTATE * DSTATE * DSTATE; idx += blockDim.x) {
        int kk = idx >> 8;
        int e  = idx & 255;
        Ahat[idx] = __expf(__expf(dt[kk]) * A[e]);
    }
    __syncthreads();

    const int b    = threadIdx.x >> 5;                 // batch = wave id
    const int lane = threadIdx.x & 31;
    const int i    = lane & 15;                        // state row
    const float* bcb = bc + (size_t)b * LSEQ * NBC;
    float* hsb       = hs + (size_t)b * LSEQ * DSTATE;

    float h = 0.f;
    for (int l = 0; l < LSEQ; ++l) {
        const int m = l & 15;
        const float* Am = Ahat + m * 256 + i * 16;
        float s = 0.f;
        #pragma unroll
        for (int j = 0; j < 16; ++j) {
            float hj = __shfl(h, j, 32);
            s += Am[j] * hj;
        }
        const float Bt = bcb[l * NBC + i];
        const float Ct = bcb[l * NBC + 16 + i];
        h = tanhf(s + Bt * h + Ct);
        if (lane < 16) hsb[l * DSTATE + lane] = h;
    }
}

// ---------------------------------------------------------------- y = tile(hs)+res
__global__ void build_y(const float* __restrict__ hs,
                        const float* __restrict__ res,
                        unsigned short* __restrict__ yb)
{
    int idx = blockIdx.x * blockDim.x + threadIdx.x;
    if (idx >= BATCH * LSEQ * DI) return;
    int c  = idx % DI;
    int ml = idx / DI;                                  // b*L + l
    float v = hs[(size_t)ml * DSTATE + (c & (DSTATE - 1))] + res[idx];
    yb[idx] = f2bf(v);
}

// ---------------------------------------------------------------- launch
extern "C" void kernel_launch(void* const* d_in, const int* in_sizes, int n_in,
                              void* d_out, int out_size, void* d_ws, size_t ws_size,
                              hipStream_t stream) {
    (void)in_sizes; (void)n_in; (void)out_size; (void)ws_size;
    const float* x      = (const float*)d_in[0];
    const float* W_in   = (const float*)d_in[1];
    const float* b_in   = (const float*)d_in[2];
    const float* conv_w = (const float*)d_in[3];
    const float* conv_b = (const float*)d_in[4];
    const float* W_x    = (const float*)d_in[5];
    const float* b_x    = (const float*)d_in[6];
    const float* dt     = (const float*)d_in[7];
    const float* A      = (const float*)d_in[8];
    /* d_in[9] = D: computed-but-unused in reference */
    const float* W_out  = (const float*)d_in[10];
    const float* b_out  = (const float*)d_in[11];
    float* out = (float*)d_out;

    const int ML = BATCH * LSEQ;                        // 8192
    char* ws = (char*)d_ws;
    size_t off = 0;
    auto carve = [&](size_t bytes) {
        void* p = ws + off;
        off += (bytes + 255) & ~(size_t)255;
        return p;
    };
    unsigned short* xb  = (unsigned short*)carve((size_t)ML * DMOD * 2);     // x bf16
    unsigned short* Wib = (unsigned short*)carve((size_t)NPROJ * DMOD * 2);  // W_in bf16
    unsigned short* Wxb = (unsigned short*)carve((size_t)NBC * DI * 2);      // W_x bf16
    unsigned short* Wob = (unsigned short*)carve((size_t)DMOD * DI * 2);     // W_out bf16
    float*          xi  = (float*)         carve((size_t)ML * DI * 4);       // conv input
    float*          res = (float*)         carve((size_t)ML * DI * 4);       // residual
    unsigned short* xsb = (unsigned short*)carve((size_t)ML * DI * 2);       // silu(conv) bf16
    float*          bc  = (float*)         carve((size_t)ML * NBC * 4);      // B,C
    float*          hs  = (float*)         carve((size_t)ML * DSTATE * 4);   // scan states
    unsigned short* yb  = (unsigned short*)carve((size_t)ML * DI * 2);       // y bf16

    // 1) bf16 conversions
    {
        int n = ML * DMOD;
        cvt_f32_to_bf16<<<(n + 255) / 256, 256, 0, stream>>>(x, xb, n);
        n = NPROJ * DMOD;
        cvt_f32_to_bf16<<<(n + 255) / 256, 256, 0, stream>>>(W_in, Wib, n);
        n = NBC * DI;
        cvt_f32_to_bf16<<<(n + 255) / 256, 256, 0, stream>>>(W_x, Wxb, n);
        n = DMOD * DI;
        cvt_f32_to_bf16<<<(n + 255) / 256, 256, 0, stream>>>(W_out, Wob, n);
    }

    // 2) GEMM1: (8192x1024)@(1024x4096)^T + b_in, split -> xi | res
    {
        int waves = (ML / 32) * (NPROJ / 64);           // 256 * 64 = 16384
        gemm_wmma_bf16<4, 1, DMOD, NPROJ><<<waves / 8, 256, 0, stream>>>(
            (const bf16_t*)xb, (const bf16_t*)Wib, b_in, xi, res, ML, DI);
    }

    // 3) depthwise causal conv + silu -> xs (bf16)
    {
        int n = ML * DI;
        conv_silu<<<(n + 255) / 256, 256, 0, stream>>>(xi, conv_w, conv_b, xsb);
    }

    // 4) GEMM2: (8192x2048)@(2048x32)^T + b_x -> bc   (N=32 -> NT=2 tiles)
    {
        int waves = (ML / 32) * (NBC / 32);             // 256 * 1 = 256
        gemm_wmma_bf16<2, 0, DI, NBC><<<waves / 8, 256, 0, stream>>>(
            (const bf16_t*)xsb, (const bf16_t*)Wxb, b_x, bc, nullptr, ML, 0);
    }

    // 5) selective scan (serial over L; one wave per batch)
    scan_kernel<<<1, 128, 0, stream>>>(bc, dt, A, hs);

    // 6) y = tile(hs, 128) + res -> bf16
    {
        int n = ML * DI;
        build_y<<<(n + 255) / 256, 256, 0, stream>>>(hs, res, yb);
    }

    // 7) GEMM3: (8192x2048)@(2048x1024)^T + b_out -> out (f32)
    {
        int waves = (ML / 32) * (DMOD / 64);            // 256 * 16 = 4096
        gemm_wmma_bf16<4, 0, DI, DMOD><<<waves / 8, 256, 0, stream>>>(
            (const bf16_t*)yb, (const bf16_t*)Wob, b_out, out, nullptr, ML, 0);
    }
}
